// EGCNet_38044820308174
// MI455X (gfx1250) — matrix operand order repaired and checked by reading, not compiled, not run on hardware
//
#include <hip/hip_runtime.h>

// ---------------------------------------------------------------------------
// MI455X (gfx1250, wave32) EGNN layer:
//   edge MLP (K=257,128 GEMMs) -> gated scatter-add -> node MLP (K=256,128 x6)
// All 128-wide GEMMs run on v_wmma_f32_16x16x32_bf16 with f32 accumulators.
// Weights are pre-packed into B-fragment layout (bf16) in d_ws; activations
// are written to LDS directly in the A-fragment swizzle. tanh is computed
// with hardware v_exp_f32/v_rcp_f32 (TRANS pipe, co-executes with WMMA).
// ---------------------------------------------------------------------------

typedef __attribute__((ext_vector_type(16))) __bf16 v16bf;
typedef __attribute__((ext_vector_type(8)))  float  v8f;

union V16U { uint4 q[2]; v16bf v; };

__device__ __forceinline__ unsigned short f2bf(float f) {
  unsigned u = __float_as_uint(f);
  u += 0x7FFFu + ((u >> 16) & 1u);          // round-to-nearest-even
  return (unsigned short)(u >> 16);
}

__device__ __forceinline__ float fast_tanh(float x) {
  // tanh(x) = (e^{2x}-1)/(e^{2x}+1); clamp keeps e^{2x} finite.
  float cx = fminf(fmaxf(x, -15.f), 15.f);
  float e2 = __expf(2.f * cx);              // v_exp_f32 (TRANS)
  return (e2 - 1.f) * __builtin_amdgcn_rcpf(e2 + 1.f);  // v_rcp_f32 (TRANS)
}

__device__ __forceinline__ v16bf ld_frag(const unsigned short* p) {
  V16U u;
  const uint4* q = (const uint4*)p;
  u.q[0] = q[0];
  u.q[1] = q[1];
  return u.v;
}

__device__ __forceinline__ v8f wmma_bf16(v16bf a, v16bf b, v8f c) {
  return __builtin_amdgcn_wmma_f32_16x16x32_bf16(false, a, false, b,
                                                 (short)0, c, false, false);
}

__device__ __forceinline__ void atom_add_f32(float* p, float v) {
  __hip_atomic_fetch_add(p, v, __ATOMIC_RELAXED, __HIP_MEMORY_SCOPE_AGENT);
}

__device__ __forceinline__ v8f vzero8() {
  v8f z = {0.f, 0.f, 0.f, 0.f, 0.f, 0.f, 0.f, 0.f};
  return z;
}

// Scatter one f32 into the bf16 A-fragment LDS swizzle.
// A 16x32 layout: lane = ((k&15)>>3)*16 + row ; elem e = (k>>4)*8 + (k&7)
__device__ __forceinline__ void st_fragA(unsigned short* buf, int rowtileC,
                                         int cbase, int row, int col, float v) {
  int chunk  = cbase + (col >> 5);
  int kin    = col & 31;
  int lane_t = ((kin >> 3) & 1) * 16 + row;
  int eo     = (kin >> 4) * 8 + (kin & 7);
  buf[((size_t)(rowtileC + chunk) * 32 + lane_t) * 16 + eo] = f2bf(v);
}

// Store 4 consecutive feats (feat%4==0) -> 4 consecutive fragment elems (8B).
__device__ __forceinline__ void st_fragA4(unsigned short* buf, int rowtileC,
                                          int cbase, int row, int feat, float4 v) {
  int chunk  = cbase + (feat >> 5);
  int kin    = feat & 31;
  int lane_t = ((kin >> 3) & 1) * 16 + row;
  int eo     = (kin >> 4) * 8 + (kin & 7);           // multiple of 4
  uint2 p = make_uint2((unsigned)f2bf(v.x) | ((unsigned)f2bf(v.y) << 16),
                       (unsigned)f2bf(v.z) | ((unsigned)f2bf(v.w) << 16));
  *(uint2*)&buf[((size_t)(rowtileC + chunk) * 32 + lane_t) * 16 + eo] = p;
}

// ---------------------------------------------------------------------------
// Weight packing: W[K][128] f32 (row-major) -> bf16 B-fragments
// dst[chunk][ct][lane][e], B 32x16 layout: col = ct*16 + (lane&15),
// k = chunk*32 + (lane>>4)*8 + (e&7) + (e>>3)*16
// ---------------------------------------------------------------------------
__global__ void pack_w_kernel(const float* __restrict__ W,
                              unsigned short* __restrict__ dst, int K) {
  int t = blockIdx.x * blockDim.x + threadIdx.x;
  int total = (K >> 5) * 8 * 32;
  if (t >= total) return;
  int lane  = t & 31;
  int ct    = (t >> 5) & 7;
  int chunk = t >> 8;
  int col   = ct * 16 + (lane & 15);
  int kbase = chunk * 32 + (lane >> 4) * 8;
  unsigned short* o = dst + ((size_t)(chunk * 8 + ct) * 32 + lane) * 16;
#pragma unroll
  for (int e = 0; e < 16; ++e)
    o[e] = f2bf(W[(size_t)(kbase + (e & 7) + (e >> 3) * 16) * 128 + col]);
}

__global__ void zero_f4_kernel(float4* p, int n4) {
  int i = blockIdx.x * blockDim.x + threadIdx.x;
  if (i < n4) p[i] = make_float4(0.f, 0.f, 0.f, 0.f);
}

// ---------------------------------------------------------------------------
// Edge kernel: 64 edges / block, 128 threads (4 waves, 1 row-tile each)
// ---------------------------------------------------------------------------
__global__ __launch_bounds__(128) void egc_edge_kernel(
    const float* __restrict__ x, const float* __restrict__ pos,
    const int* __restrict__ e_st, const int* __restrict__ e_end,
    const unsigned short* __restrict__ w1p,   // [8 chunks][8 ct][32][16]
    const float* __restrict__ w1_last,        // fe_w1 row 256
    const float* __restrict__ b1,
    const unsigned short* __restrict__ w2p,   // [4][8][32][16]
    const float* __restrict__ b2,
    const float* __restrict__ finf_w, const float* __restrict__ finf_b,
    float* __restrict__ m_i, int E) {
  __shared__ __align__(16) unsigned short fragS[4 * 4 * 32 * 16]; // x[src] / m1
  __shared__ __align__(16) unsigned short fragD[4 * 4 * 32 * 16]; // x[dst]
  __shared__ float s_dist[64];
  __shared__ int   s_src[64];
  __shared__ int   s_dst[64];

  const int tid  = threadIdx.x;
  const int lane = tid & 31;
  const int wv   = tid >> 5;                   // row-tile id 0..3
  const int e0   = blockIdx.x * 64;
  const int halfl = lane >> 4, cl = lane & 15;

  // per-edge scalars
  for (int e = tid; e < 64; e += 128) {
    int eid = e0 + e;
    if (eid < E) {
      int s = e_st[eid], d = e_end[eid];
      float dx = pos[s * 3 + 0] - pos[d * 3 + 0];
      float dy = pos[s * 3 + 1] - pos[d * 3 + 1];
      float dz = pos[s * 3 + 2] - pos[d * 3 + 2];
      // ||(pos_s - pos_d)^2||_2 = sqrt(sum d^4)
      s_dist[e] = sqrtf(dx * dx * dx * dx + dy * dy * dy * dy + dz * dz * dz * dz);
      s_src[e] = s;
      s_dst[e] = d;
    } else {
      s_dist[e] = 0.f; s_src[e] = -1; s_dst[e] = -1;
    }
  }
  __syncthreads();

  // gather x[src], x[dst] into bf16 A-fragment LDS layout
  for (int t = tid; t < 64 * 32; t += 128) {
    int er = t >> 5, g = t & 31, feat = g * 4;
    int src = s_src[er], dst = s_dst[er];
    float4 vs = make_float4(0.f, 0.f, 0.f, 0.f), vd = vs;
    if (src >= 0) {
      vs = *(const float4*)(x + (size_t)src * 128 + feat);
      vd = *(const float4*)(x + (size_t)dst * 128 + feat);
    }
    int rowtileC = (er >> 4) * 4, row_in = er & 15;
    st_fragA4(fragS, rowtileC, 0, row_in, feat, vs);
    st_fragA4(fragD, rowtileC, 0, row_in, feat, vd);
  }
  __syncthreads();

  // ---- GEMM1: [64x257] @ fe_w1 ----
  v8f acc[8];
#pragma unroll
  for (int ct = 0; ct < 8; ++ct) acc[ct] = vzero8();
#pragma unroll
  for (int c = 0; c < 4; ++c) {
    v16bf aS = ld_frag(&fragS[((size_t)(wv * 4 + c) * 32 + lane) * 16]);
    v16bf aD = ld_frag(&fragD[((size_t)(wv * 4 + c) * 32 + lane) * 16]);
#pragma unroll
    for (int ct = 0; ct < 8; ++ct) {
      v16bf bS = ld_frag(&w1p[((size_t)((c    ) * 8 + ct) * 32 + lane) * 16]);
      v16bf bD = ld_frag(&w1p[((size_t)((c + 4) * 8 + ct) * 32 + lane) * 16]);
      acc[ct] = wmma_bf16(aS, bS, acc[ct]);
      acc[ct] = wmma_bf16(aD, bD, acc[ct]);
    }
  }
  // epilogue: + dist*row256 + b1, tanh, write m1 into fragS (own row-tile only)
#pragma unroll
  for (int ct = 0; ct < 8; ++ct) {
    int col = ct * 16 + cl;
    float wc = w1_last[col], bc = b1[col];
#pragma unroll
    for (int i = 0; i < 8; ++i) {
      int row = halfl * 8 + i;
      float v = fast_tanh(acc[ct][i] + s_dist[wv * 16 + row] * wc + bc);
      st_fragA(fragS, wv * 4, 0, row, col, v);
    }
  }

  // ---- GEMM2: m1 @ fe_w2 ----
#pragma unroll
  for (int ct = 0; ct < 8; ++ct) acc[ct] = vzero8();
#pragma unroll
  for (int c = 0; c < 4; ++c) {
    v16bf a = ld_frag(&fragS[((size_t)(wv * 4 + c) * 32 + lane) * 16]);
#pragma unroll
    for (int ct = 0; ct < 8; ++ct) {
      v16bf b = ld_frag(&w2p[((size_t)(c * 8 + ct) * 32 + lane) * 16]);
      acc[ct] = wmma_bf16(a, b, acc[ct]);
    }
  }
#pragma unroll
  for (int ct = 0; ct < 8; ++ct) {
    float bc = b2[ct * 16 + cl];
#pragma unroll
    for (int i = 0; i < 8; ++i) acc[ct][i] = fast_tanh(acc[ct][i] + bc); // m_ij
  }

  // ---- gate e_ij = tanh(m_ij . finf_w + b) ----
  float p[8];
#pragma unroll
  for (int i = 0; i < 8; ++i) p[i] = 0.f;
#pragma unroll
  for (int ct = 0; ct < 8; ++ct) {
    float wf = finf_w[ct * 16 + cl];
#pragma unroll
    for (int i = 0; i < 8; ++i) p[i] += acc[ct][i] * wf;
  }
#pragma unroll
  for (int m = 1; m <= 8; m <<= 1)
#pragma unroll
    for (int i = 0; i < 8; ++i) p[i] += __shfl_xor(p[i], m, 32);

  float fb = finf_b[0];
#pragma unroll
  for (int i = 0; i < 8; ++i) {
    int er = wv * 16 + halfl * 8 + i;
    int node = s_src[er];
    if (node >= 0) {
      float g = fast_tanh(p[i] + fb);
#pragma unroll
      for (int ct = 0; ct < 8; ++ct)
        atom_add_f32(&m_i[(size_t)node * 128 + ct * 16 + cl], g * acc[ct][i]);
    }
  }
}

// ---------------------------------------------------------------------------
// Node kernel: 64 nodes / block, 128 threads (4 waves)
// ---------------------------------------------------------------------------
__global__ __launch_bounds__(128) void egc_node_kernel(
    const float* __restrict__ x, const float* __restrict__ m_i,
    const unsigned short* __restrict__ fh1p,  // [8][8][32][16]
    const float* __restrict__ fh_b1,
    const unsigned short* __restrict__ fh2p, const float* __restrict__ fh_b2,
    const unsigned short* __restrict__ l1p, const float* __restrict__ l1_b,
    const unsigned short* __restrict__ l2p, const float* __restrict__ l2_b,
    const unsigned short* __restrict__ l3p, const float* __restrict__ l3_b,
    const unsigned short* __restrict__ l4p, const float* __restrict__ l4_b,
    const float* __restrict__ l5_w, const float* __restrict__ l5_b,
    float* __restrict__ out, int N) {
  __shared__ __align__(16) unsigned short frag[4 * 8 * 32 * 16]; // [rt][8ch][32][16]

  const int tid  = threadIdx.x;
  const int lane = tid & 31;
  const int wv   = tid >> 5;
  const int n0   = blockIdx.x * 64;
  const int halfl = lane >> 4, cl = lane & 15;

  // gather: x -> chunks 0..3, m_i -> chunks 4..7
  for (int t = tid; t < 64 * 32; t += 128) {
    int nr = t >> 5, g = t & 31, feat = g * 4;
    int node = n0 + nr;
    float4 xv = make_float4(0.f, 0.f, 0.f, 0.f), mv = xv;
    if (node < N) {
      xv = *(const float4*)(x   + (size_t)node * 128 + feat);
      mv = *(const float4*)(m_i + (size_t)node * 128 + feat);
    }
    int rowtileC = (nr >> 4) * 8, row_in = nr & 15;
    st_fragA4(frag, rowtileC, 0, row_in, feat, xv);
    st_fragA4(frag, rowtileC, 4, row_in, feat, mv);
  }
  __syncthreads();

  v8f acc[8];

  // ---- fh1: [64x256] @ fh_w1, tanh -> chunks 4..7 ----
#pragma unroll
  for (int ct = 0; ct < 8; ++ct) acc[ct] = vzero8();
#pragma unroll
  for (int c = 0; c < 8; ++c) {
    v16bf a = ld_frag(&frag[((size_t)(wv * 8 + c) * 32 + lane) * 16]);
#pragma unroll
    for (int ct = 0; ct < 8; ++ct) {
      v16bf b = ld_frag(&fh1p[((size_t)(c * 8 + ct) * 32 + lane) * 16]);
      acc[ct] = wmma_bf16(a, b, acc[ct]);
    }
  }
#pragma unroll
  for (int ct = 0; ct < 8; ++ct) {
    int col = ct * 16 + cl;
    float bc = fh_b1[col];
#pragma unroll
    for (int i = 0; i < 8; ++i)
      st_fragA(frag, wv * 8, 4, halfl * 8 + i, col, fast_tanh(acc[ct][i] + bc));
  }

  // ---- fh2 + residual: x' = x + t1 @ fh_w2 + b2 -> chunks 0..3 ----
#pragma unroll
  for (int ct = 0; ct < 8; ++ct) acc[ct] = vzero8();
#pragma unroll
  for (int c = 0; c < 4; ++c) {
    v16bf a = ld_frag(&frag[((size_t)(wv * 8 + 4 + c) * 32 + lane) * 16]);
#pragma unroll
    for (int ct = 0; ct < 8; ++ct) {
      v16bf b = ld_frag(&fh2p[((size_t)(c * 8 + ct) * 32 + lane) * 16]);
      acc[ct] = wmma_bf16(a, b, acc[ct]);
    }
  }
#pragma unroll
  for (int ct = 0; ct < 8; ++ct) {
    int col = ct * 16 + cl;
    float bc = fh_b2[col];
#pragma unroll
    for (int i = 0; i < 8; ++i) {
      int row = halfl * 8 + i;
      int node = n0 + wv * 16 + row;
      float xg = (node < N) ? x[(size_t)node * 128 + col] : 0.f;
      st_fragA(frag, wv * 8, 0, row, col, xg + acc[ct][i] + bc);
    }
  }

  // ---- l1..l4 (leaky relu 0.01), ping-pong LDS fragment halves ----
  const unsigned short* lw[4] = {l1p, l2p, l3p, l4p};
  const float* lb[4] = {l1_b, l2_b, l3_b, l4_b};
#pragma unroll
  for (int layer = 0; layer < 4; ++layer) {
    int cin  = (layer & 1) ? 4 : 0;
    int cout = cin ^ 4;
#pragma unroll
    for (int ct = 0; ct < 8; ++ct) acc[ct] = vzero8();
#pragma unroll
    for (int c = 0; c < 4; ++c) {
      v16bf a = ld_frag(&frag[((size_t)(wv * 8 + cin + c) * 32 + lane) * 16]);
#pragma unroll
      for (int ct = 0; ct < 8; ++ct) {
        v16bf b = ld_frag(&lw[layer][((size_t)(c * 8 + ct) * 32 + lane) * 16]);
        acc[ct] = wmma_bf16(a, b, acc[ct]);
      }
    }
#pragma unroll
    for (int ct = 0; ct < 8; ++ct) {
      int col = ct * 16 + cl;
      float bc = lb[layer][col];
#pragma unroll
      for (int i = 0; i < 8; ++i) {
        float v = acc[ct][i] + bc;
        v = (v > 0.f) ? v : 0.01f * v;
        if (layer < 3)
          st_fragA(frag, wv * 8, cout, halfl * 8 + i, col, v);
        else
          acc[ct][i] = v;                   // keep l4 output in registers
      }
    }
  }

  // ---- l5: [64x128] @ [128x3] via lane reduction ----
  float p0[8], p1[8], p2[8];
#pragma unroll
  for (int i = 0; i < 8; ++i) { p0[i] = 0.f; p1[i] = 0.f; p2[i] = 0.f; }
#pragma unroll
  for (int ct = 0; ct < 8; ++ct) {
    int col = ct * 16 + cl;
    float w0 = l5_w[col * 3 + 0], w1 = l5_w[col * 3 + 1], w2 = l5_w[col * 3 + 2];
#pragma unroll
    for (int i = 0; i < 8; ++i) {
      float m = acc[ct][i];
      p0[i] += m * w0; p1[i] += m * w1; p2[i] += m * w2;
    }
  }
#pragma unroll
  for (int m = 1; m <= 8; m <<= 1)
#pragma unroll
    for (int i = 0; i < 8; ++i) {
      p0[i] += __shfl_xor(p0[i], m, 32);
      p1[i] += __shfl_xor(p1[i], m, 32);
      p2[i] += __shfl_xor(p2[i], m, 32);
    }
  if (cl == 0) {
    float b0 = l5_b[0], b1v = l5_b[1], b2v = l5_b[2];
#pragma unroll
    for (int i = 0; i < 8; ++i) {
      int node = n0 + wv * 16 + halfl * 8 + i;
      if (node < N) {
        out[(size_t)node * 3 + 0] = p0[i] + b0;
        out[(size_t)node * 3 + 1] = p1[i] + b1v;
        out[(size_t)node * 3 + 2] = p2[i] + b2v;
      }
    }
  }
}

// ---------------------------------------------------------------------------
extern "C" void kernel_launch(void* const* d_in, const int* in_sizes, int n_in,
                              void* d_out, int out_size, void* d_ws, size_t ws_size,
                              hipStream_t stream) {
  const float* x      = (const float*)d_in[0];
  const float* pos    = (const float*)d_in[1];
  const int*   eidx   = (const int*)d_in[2];
  const float* fe_w1  = (const float*)d_in[3];
  const float* fe_b1  = (const float*)d_in[4];
  const float* fe_w2  = (const float*)d_in[5];
  const float* fe_b2  = (const float*)d_in[6];
  const float* finf_w = (const float*)d_in[7];
  const float* finf_b = (const float*)d_in[8];
  const float* fh_w1  = (const float*)d_in[9];
  const float* fh_b1  = (const float*)d_in[10];
  const float* fh_w2  = (const float*)d_in[11];
  const float* fh_b2  = (const float*)d_in[12];
  const float* l1_w   = (const float*)d_in[13];
  const float* l1_b   = (const float*)d_in[14];
  const float* l2_w   = (const float*)d_in[15];
  const float* l2_b   = (const float*)d_in[16];
  const float* l3_w   = (const float*)d_in[17];
  const float* l3_b   = (const float*)d_in[18];
  const float* l4_w   = (const float*)d_in[19];
  const float* l4_b   = (const float*)d_in[20];
  const float* l5_w   = (const float*)d_in[21];
  const float* l5_b   = (const float*)d_in[22];

  const int N = in_sizes[0] / 128;
  const int E = in_sizes[2] / 2;

  // workspace: [m_i f32 N*128][packed bf16 weights]
  char* ws = (char*)d_ws;
  float* m_i = (float*)ws;
  size_t off = ((size_t)N * 128 * sizeof(float) + 255) & ~(size_t)255;
  unsigned short* w1p  = (unsigned short*)(ws + off); off += 256 * 128 * 2;
  unsigned short* w2p  = (unsigned short*)(ws + off); off += 128 * 128 * 2;
  unsigned short* fh1p = (unsigned short*)(ws + off); off += 256 * 128 * 2;
  unsigned short* fh2p = (unsigned short*)(ws + off); off += 128 * 128 * 2;
  unsigned short* p1   = (unsigned short*)(ws + off); off += 128 * 128 * 2;
  unsigned short* p2   = (unsigned short*)(ws + off); off += 128 * 128 * 2;
  unsigned short* p3   = (unsigned short*)(ws + off); off += 128 * 128 * 2;
  unsigned short* p4   = (unsigned short*)(ws + off); off += 128 * 128 * 2;

  int n4 = (N * 128) / 4;
  zero_f4_kernel<<<(n4 + 255) / 256, 256, 0, stream>>>((float4*)m_i, n4);

  pack_w_kernel<<<8, 256, 0, stream>>>(fe_w1, w1p, 256);  // rows 0..255 of [257,128]
  pack_w_kernel<<<4, 256, 0, stream>>>(fe_w2, w2p, 128);
  pack_w_kernel<<<8, 256, 0, stream>>>(fh_w1, fh1p, 256);
  pack_w_kernel<<<4, 256, 0, stream>>>(fh_w2, fh2p, 128);
  pack_w_kernel<<<4, 256, 0, stream>>>(l1_w, p1, 128);
  pack_w_kernel<<<4, 256, 0, stream>>>(l2_w, p2, 128);
  pack_w_kernel<<<4, 256, 0, stream>>>(l3_w, p3, 128);
  pack_w_kernel<<<4, 256, 0, stream>>>(l4_w, p4, 128);

  egc_edge_kernel<<<(E + 63) / 64, 128, 0, stream>>>(
      x, pos, eidx, eidx + E, w1p, fe_w1 + (size_t)256 * 128, fe_b1,
      w2p, fe_b2, finf_w, finf_b, m_i, E);

  egc_node_kernel<<<(N + 63) / 64, 128, 0, stream>>>(
      x, m_i, fh1p, fh_b1, fh2p, fh_b2, p1, l1_b, p2, l2_b, p3, l3_b,
      p4, l4_b, l5_w, l5_b, (float*)d_out, N);
}
